// SignMLPBert_64914135711813
// MI455X (gfx1250) — matrix operand order repaired and checked by reading, not compiled, not run on hardware
//
// MI455X (gfx1250) DeBERTa-style model, WMMA f16 pipeline, round 2.
// - 128x64 block tiles, v_wmma_f32_16x16x32_f16, 4 WMMA per wave per k-step.
// - B staged transposed in LDS so A and B fragments are 2x ds_load_b128 each.
// - A tile staged by the Tensor Data Mover (tensor_load_to_lds) with LDS row
//   padding done by the TDM pad engine (64B interval / 16B amount == 40-half rows),
//   synchronized with s_wait_tensorcnt; manual v8h path as fallback.
// - Fused attention per batch: WMMA scores -> rel gather + mask -> softmax ->
//   WMMA P@V with V staged transposed in LDS.

#include <hip/hip_runtime.h>

typedef __attribute__((ext_vector_type(16))) _Float16 v16h;
typedef __attribute__((ext_vector_type(8)))  _Float16 v8h;
typedef __attribute__((ext_vector_type(8)))  float    v8f;

#define BB 32
#define FF 64
#define LL 100
#define EE 256
#define DD 768
#define NTOK (BB*FF*LL)     // 204800
#define NFRM (BB*FF)        // 2048
#define KBIG (LL*EE + DD)   // 26368

// ---------------------------------------------------------------------------
// Tensor Data Mover (guarded: absent builtin -> manual vector staging)
// ---------------------------------------------------------------------------
#if defined(__has_builtin)
#if __has_builtin(__builtin_amdgcn_tensor_load_to_lds) && \
    __has_builtin(__builtin_amdgcn_s_wait_tensorcnt)
#define USE_TDM 1
#endif
#endif

#ifdef USE_TDM
typedef __attribute__((ext_vector_type(4))) unsigned int v4u;
typedef __attribute__((ext_vector_type(8))) int v8i_t;
typedef __attribute__((ext_vector_type(4))) int v4i_t;

__device__ inline unsigned lds_u32(const void* p) {
  return (unsigned)(unsigned long long)(__attribute__((address_space(3))) const char*)p;
}

// 2-D tile load: tileD1 rows of tileD0 2-byte elements, row stride strideElems,
// LDS rows padded by 16B after every 64B (matches [.][40] _Float16 LDS tiles).
__device__ inline void tdm_load_tile(unsigned ldsAddr, const _Float16* g,
                                     unsigned strideElems, unsigned tileD0,
                                     unsigned tileD1) {
  unsigned long long ga = (unsigned long long)(const void*)g;
  v4u g0;
  g0[0] = 1u;                                              // count=1, user D#
  g0[1] = ldsAddr;                                         // lds_addr
  g0[2] = (unsigned)ga;                                    // global_addr lo
  g0[3] = (unsigned)((ga >> 32) & 0x01FFFFFFu) | (2u << 30);  // addr hi | type=2
  const unsigned td0 = 0x7FFFFFFFu, td1 = 0x7FFFFFFFu;     // OOB effectively off
  v8i_t g1;
  // data_size=2B | pad_enable | pad_interval=16 DW (64B) | pad_amount=4 DW (16B)
  g1[0] = (int)((1u << 16) | (1u << 20) | (3u << 22) | (3u << 25));
  g1[1] = (int)((td0 & 0xFFFFu) << 16);
  g1[2] = (int)(((td0 >> 16) & 0xFFFFu) | ((td1 & 0xFFFFu) << 16));
  g1[3] = (int)(((td1 >> 16) & 0xFFFFu) | (tileD0 << 16));
  g1[4] = (int)(tileD1 & 0xFFFFu);                         // tile_dim1 (dim2=0)
  g1[5] = (int)strideElems;                                // dim0 stride lo32
  g1[6] = 0;
  g1[7] = 0;
  v4i_t z4 = {0, 0, 0, 0};
#if __clang_major__ >= 23
  v8i_t z8 = {0, 0, 0, 0, 0, 0, 0, 0};
  __builtin_amdgcn_tensor_load_to_lds(g0, g1, z4, z4, z8, 0);
#else
  __builtin_amdgcn_tensor_load_to_lds(g0, g1, z4, z4, 0);
#endif
}
__device__ inline void tdm_wait() { __builtin_amdgcn_s_wait_tensorcnt((short)0); }
#endif  // USE_TDM

// ---------------------------------------------------------------------------
// WMMA helpers (fragment layouts per CDNA5 ISA 7.12.2, wave32)
// ---------------------------------------------------------------------------
__device__ inline v8f wmma32(v16h a, v16h b, v8f c) {
  return __builtin_amdgcn_wmma_f32_16x16x32_f16(false, a, false, b, (short)0, c,
                                                false, false);
}

// A fragment from row-major LDS tile (ld halves/row, 16B-aligned rows):
// two contiguous 8-half runs -> 2x ds_load_b128.
__device__ inline v16h ld_a_frag(const _Float16* S, int ld, int row0, int kOff, int lane) {
  int r  = row0 + (lane & 15);
  int kb = kOff + ((lane < 16) ? 0 : 8);
  v8h lo = *(const v8h*)(S + (size_t)r * ld + kb);
  v8h hi = *(const v8h*)(S + (size_t)r * ld + kb + 16);
  return __builtin_shufflevector(lo, hi, 0, 1, 2, 3, 4, 5, 6, 7, 8, 9, 10, 11, 12,
                                 13, 14, 15);
}

// B fragment from an N-major (transposed) LDS tile Bt[n][k]:
// b[j] = Bt[n][kOff+ko+j] (j<8), Bt[n][kOff+ko+16+j-8] (j>=8), ko=0/8 -> 2x ds_load_b128.
__device__ inline v16h ld_bT_frag(const _Float16* S, int ld, int col0, int kOff, int lane) {
  int c  = col0 + (lane & 15);
  int ko = kOff + ((lane < 16) ? 0 : 8);
  v8h lo = *(const v8h*)(S + (size_t)c * ld + ko);
  v8h hi = *(const v8h*)(S + (size_t)c * ld + ko + 16);
  return __builtin_shufflevector(lo, hi, 0, 1, 2, 3, 4, 5, 6, 7, 8, 9, 10, 11, 12,
                                 13, 14, 15);
}

// ---------------------------------------------------------------------------
// Block reduce (blockDim.x == 256)
// ---------------------------------------------------------------------------
__device__ inline float block_reduce_sum(float v, float* sbuf) {
  int tid = threadIdx.x;
  sbuf[tid] = v;
  __syncthreads();
  for (int s = 128; s > 0; s >>= 1) {
    if (tid < s) sbuf[tid] += sbuf[tid + s];
    __syncthreads();
  }
  float r = sbuf[0];
  __syncthreads();
  return r;
}

// ---------------------------------------------------------------------------
// Generic WMMA GEMM. Block tile 128x64, K-step 32, 8 waves -> 2x2 16x16 tiles
// per wave (4 WMMA / wave / k-step). bTrans: B is row-major N x K.
// Epilogue: +bias, *scale+shift (folded BN), act (1=relu, 2=leaky 0.01),
// optional f32 out, optional f16 out, scatterL maps token rows into W1 layout.
// ---------------------------------------------------------------------------
__global__ void __launch_bounds__(256) k_gemm(
    const _Float16* __restrict__ A, const _Float16* __restrict__ B,
    int M, int N, int K, int lda, int ldb, int bTrans,
    const float* __restrict__ bias, const float* __restrict__ scale,
    const float* __restrict__ shift, int act,
    float* __restrict__ outF, int ldf,
    _Float16* __restrict__ outH, int ldh, int scatterL) {
  __shared__ alignas(16) _Float16 As[128][40];
  __shared__ alignas(16) _Float16 Bt[64][40];
  const int tid  = threadIdx.x;
  const int lane = tid & 31, wave = tid >> 5;
  const int m0 = blockIdx.y * 128, n0 = blockIdx.x * 64;
  const int mg = wave >> 1, ng = wave & 1;
  v8f acc[2][2] = {};
  const bool alA = ((((size_t)A) | ((size_t)(lda * 2))) & 15) == 0;
  const bool alB = ((((size_t)B) | ((size_t)(ldb * 2))) & 15) == 0;

  for (int k0 = 0; k0 < K; k0 += 32) {
    const bool fullK = (k0 + 32 <= K);
    const bool fastA = alA && fullK && (m0 + 128 <= M);
    // ---- stage A (128 x 32)
    if (fastA) {
#ifdef USE_TDM
      if (wave == 0)
        tdm_load_tile(lds_u32(&As[0][0]), A + (size_t)m0 * lda + k0, (unsigned)lda,
                      32u, 128u);
#else
#pragma unroll
      for (int t = 0; t < 2; ++t) {
        int ch = tid + t * 256;
        int r = ch >> 2, c8 = (ch & 3) * 8;
        *(v8h*)&As[r][c8] = *(const v8h*)&A[(size_t)(m0 + r) * lda + k0 + c8];
      }
#endif
    } else {
      for (int idx = tid; idx < 128 * 32; idx += 256) {
        int r = idx >> 5, c = idx & 31;
        int gm = m0 + r, gk = k0 + c;
        As[r][c] = (gm < M && gk < K) ? A[(size_t)gm * lda + gk] : (_Float16)0.f;
      }
    }
    // ---- stage B transposed into Bt[n][k]
    if (bTrans) {
      if (alB && fullK && (n0 + 64 <= N)) {
        int n = tid >> 2, kc = (tid & 3) * 8;
        *(v8h*)&Bt[n][kc] = *(const v8h*)&B[(size_t)(n0 + n) * ldb + k0 + kc];
      } else {
        for (int idx = tid; idx < 64 * 32; idx += 256) {
          int n = idx >> 5, k = idx & 31;
          int gn = n0 + n, gk = k0 + k;
          Bt[n][k] = (gn < N && gk < K) ? B[(size_t)gn * ldb + gk] : (_Float16)0.f;
        }
      }
    } else {
      if (alB && (n0 + 64 <= N)) {
        int k = tid >> 3, nc = (tid & 7) * 8;
        int gk = k0 + k;
        v8h bv = {};
        if (gk < K) bv = *(const v8h*)&B[(size_t)gk * ldb + n0 + nc];
        if (gk + 32 < K)
          __builtin_prefetch(&B[(size_t)(gk + 32) * ldb + n0 + nc], 0, 1);
#pragma unroll
        for (int j = 0; j < 8; ++j) Bt[nc + j][k] = bv[j];
      } else {
        for (int idx = tid; idx < 64 * 32; idx += 256) {
          int n = idx >> 5, k = idx & 31;
          int gn = n0 + n, gk = k0 + k;
          Bt[n][k] = (gn < N && gk < K) ? B[(size_t)gk * ldb + gn] : (_Float16)0.f;
        }
      }
    }
#ifdef USE_TDM
    if (fastA && wave == 0) tdm_wait();
#endif
    __syncthreads();
    // ---- compute
    v16h a0 = ld_a_frag(&As[0][0], 40, mg * 32, 0, lane);
    v16h a1 = ld_a_frag(&As[0][0], 40, mg * 32 + 16, 0, lane);
    v16h b0 = ld_bT_frag(&Bt[0][0], 40, ng * 32, 0, lane);
    v16h b1 = ld_bT_frag(&Bt[0][0], 40, ng * 32 + 16, 0, lane);
    acc[0][0] = wmma32(a0, b0, acc[0][0]);
    acc[0][1] = wmma32(a0, b1, acc[0][1]);
    acc[1][0] = wmma32(a1, b0, acc[1][0]);
    acc[1][1] = wmma32(a1, b1, acc[1][1]);
    __syncthreads();
  }

  // ---- epilogue (C layout: VGPR r -> M=r / 8+r; N = lane&15)
  const int rowAdj = (lane & 16) ? 8 : 0;
  const int cl = lane & 15;
  for (int im = 0; im < 2; ++im) {
    for (int in = 0; in < 2; ++in) {
      int gn = n0 + ng * 32 + in * 16 + cl;
      if (gn >= N) continue;
#pragma unroll
      for (int r = 0; r < 8; ++r) {
        int gm = m0 + mg * 32 + im * 16 + rowAdj + r;
        if (gm >= M) continue;
        float v = acc[im][in][r];
        if (bias)  v += bias[gn];
        if (scale) v = v * scale[gn] + shift[gn];
        if (act == 1)      v = v > 0.f ? v : 0.f;
        else if (act == 2) v = v > 0.f ? v : 0.01f * v;
        if (outF) outF[(size_t)gm * ldf + gn] = v;
        if (outH) {
          size_t oi = scatterL ? ((size_t)(gm / LL) * ldh + (size_t)(gm % LL) * EE + gn)
                               : ((size_t)gm * ldh + gn);
          outH[oi] = (_Float16)v;
        }
      }
    }
  }
}

// ---------------------------------------------------------------------------
// Fused DeBERTa attention, one block (8 waves) per batch. S=64, DH=768, H=1.
// ---------------------------------------------------------------------------
__global__ void __launch_bounds__(256) k_attn(
    const _Float16* __restrict__ q16, const _Float16* __restrict__ k16,
    const _Float16* __restrict__ v16, const float* __restrict__ c2pf,
    const float* __restrict__ p2cf, const int* __restrict__ c2pi,
    const int* __restrict__ p2ci, const float* __restrict__ mask,
    _Float16* __restrict__ ctx16) {
  __shared__ alignas(16) _Float16 qs[64][40];
  __shared__ alignas(16) _Float16 ks[64][40];
  __shared__ alignas(16) float    sS[64][65];
  __shared__ alignas(16) _Float16 p16[64][72];
  __shared__ alignas(16) _Float16 vt[DD][72];   // V transposed: vt[n][k]
  const int tid = threadIdx.x, lane = tid & 31, wave = tid >> 5;
  const int bb  = blockIdx.x;
  const size_t base = (size_t)bb * 64;
  const int mi = wave >> 1, njb = (wave & 1) * 2;
  v8f acc0 = {}, acc1 = {};

  // stage V transposed (64 x 768 -> vt[768][72]); consumed after later barriers
#pragma unroll
  for (int t = 0; t < 24; ++t) {
    int ch = tid + t * 256;           // 6144 chunks of 8 halves
    int k = ch / 96, nc = (ch % 96) * 8;
    v8h vv = *(const v8h*)&v16[(base + k) * DD + nc];
#pragma unroll
    for (int j = 0; j < 8; ++j) vt[nc + j][k] = vv[j];
  }

  // scores = q @ k^T  (K = 768)
  for (int k0 = 0; k0 < DD; k0 += 32) {
    int r = tid >> 2, kc = (tid & 3) * 8;
    *(v8h*)&qs[r][kc] = *(const v8h*)&q16[(base + r) * DD + k0 + kc];
    *(v8h*)&ks[r][kc] = *(const v8h*)&k16[(base + r) * DD + k0 + kc];
    __syncthreads();
    v16h af = ld_a_frag(&qs[0][0], 40, mi * 16, 0, lane);
    v16h b0 = ld_bT_frag(&ks[0][0], 40, njb * 16, 0, lane);
    v16h b1 = ld_bT_frag(&ks[0][0], 40, (njb + 1) * 16, 0, lane);
    acc0 = wmma32(af, b0, acc0);
    acc1 = wmma32(af, b1, acc1);
    __syncthreads();
  }
  const int rowOff = mi * 16 + ((lane & 16) ? 8 : 0);
  const int cl = lane & 15;
#pragma unroll
  for (int r = 0; r < 8; ++r) {
    sS[rowOff + r][njb * 16 + cl]       = acc0[r];
    sS[rowOff + r][(njb + 1) * 16 + cl] = acc1[r];
  }
  __syncthreads();

  // rel-position terms + mask
  const float scl = 1.0f / 48.0f;  // 1/sqrt(DH*3)
  for (int e = tid; e < 4096; e += 256) {
    int i = e >> 6, j = e & 63;
    float s = sS[i][j] * scl;
    s += (c2pf[(base + i) * 100 + c2pi[i * 64 + j]] +
          p2cf[(base + j) * 100 + p2ci[j * 64 + i]]) * scl;
    bool ok = (mask[base + i] * mask[base + j]) > 0.f;
    sS[i][j] = ok ? s : -3.4028235e38f;
  }
  __syncthreads();

  // row softmax -> f16 probs
  if (tid < 64) {
    float mx = -3.4028235e38f;
    for (int j = 0; j < 64; ++j) mx = fmaxf(mx, sS[tid][j]);
    float sum = 0.f;
    for (int j = 0; j < 64; ++j) { float e2 = __expf(sS[tid][j] - mx); sS[tid][j] = e2; sum += e2; }
    float inv = 1.f / sum;
    for (int j = 0; j < 64; ++j) p16[tid][j] = (_Float16)(sS[tid][j] * inv);
  }
  __syncthreads();

  // ctx = P(64x64) @ V(64x768); 4x48 tiles of 16x16 over 8 waves
  for (int t = wave; t < 4 * 48; t += 8) {
    int tmi = t / 48, tnj = t % 48;
    v8f acc = {};
    for (int ks0 = 0; ks0 < 64; ks0 += 32) {
      v16h af = ld_a_frag(&p16[0][0], 72, tmi * 16, ks0, lane);
      v16h bf = ld_bT_frag(&vt[0][0], 72, tnj * 16, ks0, lane);
      acc = wmma32(af, bf, acc);
    }
#pragma unroll
    for (int r = 0; r < 8; ++r) {
      int gm = tmi * 16 + ((lane & 16) ? 8 : 0) + r;
      ctx16[(base + gm) * DD + tnj * 16 + cl] = (_Float16)acc[r];
    }
  }
}

// ---------------------------------------------------------------------------
// Elementwise / small kernels
// ---------------------------------------------------------------------------
__global__ void k_cvt16(const float* __restrict__ x, _Float16* __restrict__ y, size_t n) {
  for (size_t i = blockIdx.x * (size_t)blockDim.x + threadIdx.x; i < n;
       i += (size_t)gridDim.x * blockDim.x)
    y[i] = (_Float16)x[i];
}

// Token row of Xcat: [ LN(type_emb) | LN(lmk_emb) | add_shift(pos3) @ Wpos ]
__global__ void __launch_bounds__(256) k_embed(
    const float* __restrict__ x, const float* __restrict__ y, const float* __restrict__ z,
    const int* __restrict__ tids, const int* __restrict__ lids,
    const float* __restrict__ temb, const float* __restrict__ lemb,
    const float* __restrict__ tg, const float* __restrict__ tb,
    const float* __restrict__ lg, const float* __restrict__ lb,
    const float* __restrict__ Wpos, _Float16* __restrict__ Xcat) {
  __shared__ float sbuf[256];
  const int t = blockIdx.x, tid = threadIdx.x;
  const int l = t % LL, bf = t / LL;
  const int f = bf % FF, b = bf / FF;
  size_t o = (size_t)t * DD;

  float e = temb[(size_t)tids[t] * EE + tid];
  float m = block_reduce_sum(e, sbuf) * (1.f / EE);
  float d = e - m;
  float v = block_reduce_sum(d * d, sbuf) * (1.f / EE);
  Xcat[o + tid] = (_Float16)(d * rsqrtf(v + 1e-5f) * tg[tid] + tb[tid]);

  e = lemb[(size_t)lids[t] * EE + tid];
  m = block_reduce_sum(e, sbuf) * (1.f / EE);
  d = e - m;
  v = block_reduce_sum(d * d, sbuf) * (1.f / EE);
  Xcat[o + EE + tid] = (_Float16)(d * rsqrtf(v + 1e-5f) * lg[tid] + lb[tid]);

  // add_shift along frame axis F
  int fp = f > 0 ? f - 1 : 0, fn = f < FF - 1 ? f + 1 : FF - 1;
  size_t t0 = t, t1 = ((size_t)b * FF + fp) * LL + l, t2 = ((size_t)b * FF + fn) * LL + l;
  float p[9] = {x[t0], y[t0], z[t0], x[t1], y[t1], z[t1], x[t2], y[t2], z[t2]};
  float acc = 0.f;
#pragma unroll
  for (int j = 0; j < 9; ++j) acc += p[j] * Wpos[j * EE + tid];
  Xcat[o + 2 * EE + tid] = (_Float16)acc;
}

__global__ void __launch_bounds__(256) k_pdist(
    const float* __restrict__ x, const float* __restrict__ y, const float* __restrict__ z,
    _Float16* __restrict__ pd) {
  const int bf = blockIdx.x;
  const size_t base = (size_t)bf * LL;
  for (int idx = threadIdx.x; idx < 882; idx += 256) {
    int set = idx / 441, p = idx % 441;
    int li = set * 21 + p / 21, lj = set * 21 + p % 21;
    float dx = x[base + li] - x[base + lj];
    float dy = y[base + li] - y[base + lj];
    float dz = z[base + li] - z[base + lj];
    pd[(size_t)bf * 882 + idx] = (_Float16)sqrtf(dx * dx + dy * dy + dz * dz + 1e-8f);
  }
}

__global__ void __launch_bounds__(256) k_adjshift(const float* __restrict__ adjs,
                                                  _Float16* __restrict__ Xbig) {
  const int bf = blockIdx.x, tid = threadIdx.x;
  const int b = bf / FF, f = bf % FF;
  const int fp = f > 0 ? f - 1 : 0, fn = f < FF - 1 ? f + 1 : FF - 1;
  size_t o = (size_t)bf * KBIG + (size_t)LL * EE;
  Xbig[o + tid]          = (_Float16)adjs[((size_t)b * FF + f) * EE + tid];
  Xbig[o + EE + tid]     = (_Float16)adjs[((size_t)b * FF + fp) * EE + tid];
  Xbig[o + 2 * EE + tid] = (_Float16)adjs[((size_t)b * FF + fn) * EE + tid];
}

__global__ void k_relidx(int* __restrict__ c2pi, int* __restrict__ p2ci) {
  for (int e = threadIdx.x; e < 64 * 64; e += blockDim.x) {
    int i = e >> 6, j = e & 63;
    int rel = i - j;
    int sgn = (rel > 0) - (rel < 0);
    float abs_pos = (rel < 25 && rel > -25) ? 24.f : fabsf((float)rel);
    float log_pos = ceilf(logf(abs_pos / 25.f) / logf(99.f / 25.f) * 24.f) + 25.f;
    int bucket = (abs_pos <= 25.f) ? rel : (int)(log_pos * (float)sgn);
    int c = bucket + 50;  c = c < 0 ? 0 : (c > 99 ? 99 : c);
    int p = -bucket + 50; p = p < 0 ? 0 : (p > 99 ? 99 : p);
    c2pi[e] = c;
    p2ci[e] = p;
  }
}

__global__ void k_bn_prep(const float* __restrict__ lb, const float* __restrict__ g,
                          const float* __restrict__ bb, const float* __restrict__ m,
                          const float* __restrict__ v, float* __restrict__ scale,
                          float* __restrict__ shift, int n) {
  int i = blockIdx.x * blockDim.x + threadIdx.x;
  if (i < n) {
    float s = g[i] * rsqrtf(v[i] + 1e-5f);
    scale[i] = s;
    shift[i] = (lb[i] - m[i]) * s + bb[i];
  }
}

__global__ void __launch_bounds__(256) k_ln_row768(const float* __restrict__ x,
                                                   const float* __restrict__ g,
                                                   const float* __restrict__ bt, float eps,
                                                   _Float16* __restrict__ o16) {
  __shared__ float sbuf[256];
  const int r = blockIdx.x, tid = threadIdx.x;
  const float* xr = x + (size_t)r * DD;
  float v0 = xr[tid], v1 = xr[tid + 256], v2 = xr[tid + 512];
  float m = block_reduce_sum(v0 + v1 + v2, sbuf) * (1.f / DD);
  float d0 = v0 - m, d1 = v1 - m, d2 = v2 - m;
  float var = block_reduce_sum(d0 * d0 + d1 * d1 + d2 * d2, sbuf) * (1.f / DD);
  float rs = rsqrtf(var + eps);
  size_t o = (size_t)r * DD;
  o16[o + tid]       = (_Float16)(d0 * rs * g[tid] + bt[tid]);
  o16[o + 256 + tid] = (_Float16)(d1 * rs * g[tid + 256] + bt[tid + 256]);
  o16[o + 512 + tid] = (_Float16)(d2 * rs * g[tid + 512] + bt[tid + 512]);
}

__global__ void __launch_bounds__(256) k_add_ln(const float* __restrict__ a,
                                                float* __restrict__ h, _Float16* __restrict__ h16,
                                                const float* __restrict__ g,
                                                const float* __restrict__ bt, float eps) {
  __shared__ float sbuf[256];
  const int r = blockIdx.x, tid = threadIdx.x;
  const float* ar = a + (size_t)r * DD;
  float* hr = h + (size_t)r * DD;
  float v0 = ar[tid] + hr[tid];
  float v1 = ar[tid + 256] + hr[tid + 256];
  float v2 = ar[tid + 512] + hr[tid + 512];
  float m = block_reduce_sum(v0 + v1 + v2, sbuf) * (1.f / DD);
  float d0 = v0 - m, d1 = v1 - m, d2 = v2 - m;
  float var = block_reduce_sum(d0 * d0 + d1 * d1 + d2 * d2, sbuf) * (1.f / DD);
  float rs = rsqrtf(var + eps);
  float o0 = d0 * rs * g[tid] + bt[tid];
  float o1 = d1 * rs * g[tid + 256] + bt[tid + 256];
  float o2 = d2 * rs * g[tid + 512] + bt[tid + 512];
  hr[tid] = o0; hr[tid + 256] = o1; hr[tid + 512] = o2;
  size_t o = (size_t)r * DD;
  h16[o + tid] = (_Float16)o0;
  h16[o + 256 + tid] = (_Float16)o1;
  h16[o + 512 + tid] = (_Float16)o2;
}

__global__ void __launch_bounds__(256) k_pool(const float* __restrict__ h,
                                              const float* __restrict__ mask,
                                              _Float16* __restrict__ pooled) {
  const int b = blockIdx.x, tid = threadIdx.x;
  float denom = 0.f;
  for (int f = 0; f < FF; ++f) denom += mask[b * FF + f];
  for (int c = tid; c < DD; c += 256) {
    float s = 0.f;
    for (int f = 0; f < FF; ++f) s += h[((size_t)b * FF + f) * DD + c] * mask[b * FF + f];
    pooled[(size_t)b * DD + c] = (_Float16)(s / denom);
  }
}

// ---------------------------------------------------------------------------
// Orchestration
// ---------------------------------------------------------------------------
extern "C" void kernel_launch(void* const* d_in, const int* in_sizes, int n_in,
                              void* d_out, int out_size, void* d_ws, size_t ws_size,
                              hipStream_t stream) {
  (void)in_sizes; (void)n_in; (void)out_size; (void)ws_size;
  const float* x   = (const float*)d_in[0];
  const float* y   = (const float*)d_in[1];
  const float* z   = (const float*)d_in[2];
  const int* tids  = (const int*)d_in[3];
  const int* lids  = (const int*)d_in[4];
  const float* mask = (const float*)d_in[5];
  auto P = [&](int i) { return (const float*)d_in[6 + i]; };
  const float *type_emb = P(0), *lmk_emb = P(1), *tln_g = P(2), *tln_b = P(3),
              *lln_g = P(4), *lln_b = P(5), *Wpos = P(6), *Wd = P(7), *bd = P(8),
              *Wadj = P(9), *badj = P(10), *W1 = P(11), *b1 = P(12), *bn1_g = P(13),
              *bn1_b = P(14), *bn1_m = P(15), *bn1_v = P(16), *W2 = P(17), *b2 = P(18),
              *bn2_g = P(19), *bn2_b = P(20), *bn2_m = P(21), *bn2_v = P(22);
  const int LAYER0 = 23;  // Wq,bq,Wk,bk,Wv,bv,Wao,bao,ln1_g,ln1_b,Wi,bi,Wfo,bfo,ln2_g,ln2_b
  const float* rel_emb = P(LAYER0 + 64 + 0);
  const float* rln_g   = P(LAYER0 + 64 + 1);
  const float* rln_b   = P(LAYER0 + 64 + 2);
  const float* Wl      = P(LAYER0 + 64 + 3);
  const float* bl      = P(LAYER0 + 64 + 4);

  char* wsb = (char*)d_ws;
  size_t cur = 0;
  auto alloc = [&](size_t bytes) -> void* {
    cur = (cur + 255) & ~(size_t)255;
    void* p = wsb + cur;
    cur += bytes;
    return p;
  };
  auto a16 = [&](size_t n) { return (_Float16*)alloc(n * sizeof(_Float16)); };
  auto a32 = [&](size_t n) { return (float*)alloc(n * sizeof(float)); };

  _Float16* Wd16   = a16((size_t)DD * EE);
  _Float16* Wadj16 = a16((size_t)882 * EE);
  _Float16* W116   = a16((size_t)KBIG * 2 * DD);
  _Float16* W216   = a16((size_t)2 * DD * DD);
  _Float16* Wl16   = a16((size_t)DD * 250);
  _Float16* LW16[4][6];
  for (int li = 0; li < 4; ++li)
    for (int j = 0; j < 6; ++j) LW16[li][j] = a16((size_t)DD * DD);
  _Float16* rel16 = a16((size_t)100 * DD);
  float* scale1 = a32(2 * DD);  float* shift1 = a32(2 * DD);
  float* scale2 = a32(DD);      float* shift2 = a32(DD);
  int* c2pi = (int*)alloc(4096 * sizeof(int));
  int* p2ci = (int*)alloc(4096 * sizeof(int));
  _Float16* pd16 = a16((size_t)NFRM * 882);
  float* adjs = a32((size_t)NFRM * EE);
  _Float16* posk16 = a16((size_t)100 * DD);
  _Float16* posq16 = a16((size_t)100 * DD);
  float* c2pf = a32((size_t)NFRM * 100);
  float* p2cf = a32((size_t)NFRM * 100);
  float* tmp32 = a32((size_t)NFRM * DD);
  float* h32 = a32((size_t)NFRM * DD);
  _Float16* h16 = a16((size_t)NFRM * DD);
  _Float16* q16 = a16((size_t)NFRM * DD);
  _Float16* kk16 = a16((size_t)NFRM * DD);
  _Float16* vv16 = a16((size_t)NFRM * DD);
  _Float16* ctx16 = a16((size_t)NFRM * DD);
  _Float16* ff16 = a16((size_t)NFRM * DD);
  _Float16* fts1 = a16((size_t)NFRM * 2 * DD);
  _Float16* pooled16 = a16((size_t)BB * DD);
  _Float16* Xbig = a16((size_t)NFRM * KBIG);
  _Float16* Xcat = a16((size_t)NTOK * DD);

  auto cvt = [&](const float* s, _Float16* d, size_t n) {
    int blocks = (int)((n + 1023) / 1024);
    if (blocks > 4096) blocks = 4096;
    k_cvt16<<<blocks, 256, 0, stream>>>(s, d, n);
  };
  auto gemm = [&](const _Float16* A, const _Float16* Bm, int M, int N, int K, int lda,
                  int ldb, int bT, const float* bias, const float* sc, const float* sh,
                  int act, float* oF, int ldf, _Float16* oH, int ldh, int scat) {
    dim3 grid((N + 63) / 64, (M + 127) / 128);
    k_gemm<<<grid, 256, 0, stream>>>(A, Bm, M, N, K, lda, ldb, bT, bias, sc, sh, act,
                                     oF, ldf, oH, ldh, scat);
  };

  // weight conversions & constants
  cvt(Wd, Wd16, (size_t)DD * EE);
  cvt(Wadj, Wadj16, (size_t)882 * EE);
  cvt(W1, W116, (size_t)KBIG * 2 * DD);
  cvt(W2, W216, (size_t)2 * DD * DD);
  cvt(Wl, Wl16, (size_t)DD * 250);
  for (int li = 0; li < 4; ++li) {
    const int base = LAYER0 + li * 16;
    cvt(P(base + 0),  LW16[li][0], (size_t)DD * DD);  // Wq
    cvt(P(base + 2),  LW16[li][1], (size_t)DD * DD);  // Wk
    cvt(P(base + 4),  LW16[li][2], (size_t)DD * DD);  // Wv
    cvt(P(base + 6),  LW16[li][3], (size_t)DD * DD);  // Wao
    cvt(P(base + 10), LW16[li][4], (size_t)DD * DD);  // Wi
    cvt(P(base + 12), LW16[li][5], (size_t)DD * DD);  // Wfo
  }
  k_relidx<<<1, 256, 0, stream>>>(c2pi, p2ci);
  k_bn_prep<<<(2 * DD + 255) / 256, 256, 0, stream>>>(b1, bn1_g, bn1_b, bn1_m, bn1_v,
                                                      scale1, shift1, 2 * DD);
  k_bn_prep<<<(DD + 255) / 256, 256, 0, stream>>>(b2, bn2_g, bn2_b, bn2_m, bn2_v,
                                                  scale2, shift2, DD);
  k_ln_row768<<<100, 256, 0, stream>>>(rel_emb, rln_g, rln_b, 1e-7f, rel16);

  // feature stage
  k_embed<<<NTOK, 256, 0, stream>>>(x, y, z, tids, lids, type_emb, lmk_emb, tln_g, tln_b,
                                    lln_g, lln_b, Wpos, Xcat);
  k_pdist<<<NFRM, 256, 0, stream>>>(x, y, z, pd16);
  gemm(Xcat, Wd16, NTOK, EE, DD, DD, EE, 0, bd, nullptr, nullptr, 0,
       nullptr, 0, Xbig, KBIG, 1);
  gemm(pd16, Wadj16, NFRM, EE, 882, 882, EE, 0, badj, nullptr, nullptr, 0,
       adjs, EE, nullptr, 0, 0);
  k_adjshift<<<NFRM, 256, 0, stream>>>(adjs, Xbig);
  gemm(Xbig, W116, NFRM, 2 * DD, KBIG, KBIG, 2 * DD, 0, nullptr, scale1, shift1, 2,
       nullptr, 0, fts1, 2 * DD, 0);
  gemm(fts1, W216, NFRM, DD, 2 * DD, 2 * DD, DD, 0, nullptr, scale2, shift2, 2,
       h32, DD, h16, DD, 0);

  // transformer layers
  for (int li = 0; li < 4; ++li) {
    const int base = LAYER0 + li * 16;
    const float *bq = P(base + 1), *bk = P(base + 3), *bv = P(base + 5), *bao = P(base + 7);
    const float *ln1g = P(base + 8), *ln1b = P(base + 9);
    const float *bi = P(base + 11), *bfo = P(base + 13);
    const float *ln2g = P(base + 14), *ln2b = P(base + 15);
    gemm(rel16, LW16[li][1], 100, DD, DD, DD, DD, 0, bk, nullptr, nullptr, 0,
         nullptr, 0, posk16, DD, 0);
    gemm(rel16, LW16[li][0], 100, DD, DD, DD, DD, 0, bq, nullptr, nullptr, 0,
         nullptr, 0, posq16, DD, 0);
    gemm(h16, LW16[li][0], NFRM, DD, DD, DD, DD, 0, bq, nullptr, nullptr, 0,
         nullptr, 0, q16, DD, 0);
    gemm(h16, LW16[li][1], NFRM, DD, DD, DD, DD, 0, bk, nullptr, nullptr, 0,
         nullptr, 0, kk16, DD, 0);
    gemm(h16, LW16[li][2], NFRM, DD, DD, DD, DD, 0, bv, nullptr, nullptr, 0,
         nullptr, 0, vv16, DD, 0);
    gemm(q16, posk16, NFRM, 100, DD, DD, DD, 1, nullptr, nullptr, nullptr, 0,
         c2pf, 100, nullptr, 0, 0);
    gemm(kk16, posq16, NFRM, 100, DD, DD, DD, 1, nullptr, nullptr, nullptr, 0,
         p2cf, 100, nullptr, 0, 0);
    k_attn<<<BB, 256, 0, stream>>>(q16, kk16, vv16, c2pf, p2cf, c2pi, p2ci, mask, ctx16);
    gemm(ctx16, LW16[li][3], NFRM, DD, DD, DD, DD, 0, bao, nullptr, nullptr, 0,
         tmp32, DD, nullptr, 0, 0);
    k_add_ln<<<NFRM, 256, 0, stream>>>(tmp32, h32, h16, ln1g, ln1b, 1e-7f);
    gemm(h16, LW16[li][4], NFRM, DD, DD, DD, DD, 0, bi, nullptr, nullptr, 1,
         nullptr, 0, ff16, DD, 0);
    gemm(ff16, LW16[li][5], NFRM, DD, DD, DD, DD, 0, bfo, nullptr, nullptr, 0,
         tmp32, DD, nullptr, 0, 0);
    k_add_ln<<<NFRM, 256, 0, stream>>>(tmp32, h32, h16, ln2g, ln2b, 1e-7f);
  }

  // pool + classifier
  k_pool<<<BB, 256, 0, stream>>>(h32, mask, pooled16);
  gemm(pooled16, Wl16, BB, 250, DD, DD, 250, 0, bl, nullptr, nullptr, 0,
       (float*)d_out, 250, nullptr, 0, 0);
}
